// ObjectAttention_40278203302436
// MI455X (gfx1250) — matrix operand order repaired and checked by reading, not compiled, not run on hardware
//
#include <hip/hip_runtime.h>

typedef __bf16 bf16;
typedef __attribute__((ext_vector_type(16))) __bf16 v16bf;
typedef __attribute__((ext_vector_type(8)))  __bf16 v8bf;
typedef __attribute__((ext_vector_type(4)))  __bf16 v4bf;
typedef __attribute__((ext_vector_type(8)))  float  v8f;

#define CCH   256   // channels (M and K of conv GEMMs, feature dim of attention)
#define NPIX  9216  // 96*96
#define NBATCH 2

__device__ __forceinline__ v8f wmma_bf16(v16bf a, v16bf b, v8f c) {
  // D = A(16x32 bf16) * B(32x16 bf16) + C(16x16 f32)
  return __builtin_amdgcn_wmma_f32_16x16x32_bf16(false, a, false, b, (short)0, c,
                                                 false, false);
}

__device__ __forceinline__ v16bf combine16(v8bf lo, v8bf hi) {
  v16bf r;
#pragma unroll
  for (int i = 0; i < 8; ++i) { r[i] = lo[i]; r[i + 8] = hi[i]; }
  return r;
}
__device__ __forceinline__ v16bf load16(const bf16* p) { return *(const v16bf*)p; }
__device__ __forceinline__ v8bf  load8 (const bf16* p) { return *(const v8bf*)p; }

// ---------------------------------------------------------------------------
// Prep: fold BN params, convert weights f32 -> bf16
// ---------------------------------------------------------------------------
__global__ void prep_kernel(const float* __restrict__ Ws, const float* __restrict__ g,
                            const float* __restrict__ be, const float* __restrict__ me,
                            const float* __restrict__ va, bf16* __restrict__ Wbf,
                            float* __restrict__ scale, float* __restrict__ shift) {
  int i = blockIdx.x * blockDim.x + threadIdx.x;
  if (i < 6 * 256 * 256) Wbf[i] = (bf16)Ws[i];
  if (i < 6 * 256) {
    float s = g[i] * rsqrtf(va[i] + 1e-5f);
    scale[i] = s;
    shift[i] = be[i] - me[i] * s;
  }
}

__global__ void cvt_kernel(const float* __restrict__ src, bf16* __restrict__ dst, int n) {
  int i = blockIdx.x * blockDim.x + threadIdx.x;
  if (i < n) dst[i] = (bf16)src[i];
}

// ---------------------------------------------------------------------------
// Conv(1x1) + BN + ReLU as WMMA GEMM:  Y[o,n] = relu(sum_c W[o,c]*X[c,n]*scale+shift)
// Block: 256 threads = 8 waves arranged 2(M) x 4(N); block tile 32(M) x 64(N).
// MODE 0: Y bf16 [C,N]   MODE 1: Y bf16 [N,C]   MODE 2: Y f32 [C,N] (final out)
// ---------------------------------------------------------------------------
template <int MODE>
__global__ __launch_bounds__(256)
void convbn_kernel(const bf16* __restrict__ X, const bf16* __restrict__ Wl,
                   const float* __restrict__ scale, const float* __restrict__ shift,
                   bf16* __restrict__ Yb, float* __restrict__ Yf) {
  __shared__ __align__(32) bf16 sA[32][32];  // W tile  [o][k]
  __shared__ __align__(32) bf16 sB[64][32];  // X tile transposed [n][k]

  const int tid  = threadIdx.x;
  const int lane = tid & 31;
  const int w    = tid >> 5;
  const int l16  = lane & 15;
  const int hi   = lane >> 4;
  const int b    = blockIdx.z;
  const int oBase = blockIdx.y * 32;
  const int nBase = blockIdx.x * 64;
  const int wM = w & 1, wN = w >> 1;
  const int o0 = oBase + wM * 16;
  const int n0 = nBase + wN * 16;

  const bf16* Xb = X + (size_t)b * CCH * NPIX;

  v8f acc = {};
  const int arow = tid >> 3, acol = (tid & 7) * 4;  // A stage: 32x32, 4 elems/thread
  const int brow = tid >> 3, bcol = (tid & 7) * 8;  // B stage: 32x64, 8 elems/thread

  for (int k0 = 0; k0 < CCH; k0 += 32) {
    *(v4bf*)&sA[arow][acol] =
        *(const v4bf*)(Wl + (size_t)(oBase + arow) * CCH + k0 + acol);
    v8bf xv = load8(Xb + (size_t)(k0 + brow) * NPIX + nBase + bcol);
#pragma unroll
    for (int j = 0; j < 8; ++j) sB[bcol + j][brow] = xv[j];
    __syncthreads();

    v16bf a = load16(&sA[wM * 16 + l16][hi * 16]);
    v16bf bb = load16(&sB[wN * 16 + l16][hi * 16]);
    acc = wmma_bf16(a, bb, acc);
    __syncthreads();
  }

  const int n = n0 + l16;
  if (MODE == 1) {
    v8bf ov;
#pragma unroll
    for (int r = 0; r < 8; ++r) {
      int o = o0 + hi * 8 + r;
      float val = acc[r] * scale[o] + shift[o];
      ov[r] = (bf16)(val > 0.f ? val : 0.f);
    }
    *(v8bf*)(Yb + (size_t)b * NPIX * CCH + (size_t)n * CCH + o0 + hi * 8) = ov;
  } else {
#pragma unroll
    for (int r = 0; r < 8; ++r) {
      int o = o0 + hi * 8 + r;
      float val = acc[r] * scale[o] + shift[o];
      val = val > 0.f ? val : 0.f;
      size_t idx = (size_t)b * CCH * NPIX + (size_t)o * NPIX + n;
      if (MODE == 0) Yb[idx] = (bf16)val;
      else           Yf[idx] = val;
    }
  }
}

// ---------------------------------------------------------------------------
// Flash attention (no 1/sqrt(d) scale, matching reference).
// Q [N,K] bf16, Km [N,K] bf16 (phi output transposed), V [K,N] bf16.
// Ctx written bf16 [K,N] for the final conv.
// Block = 8 waves; each wave owns 16 query rows; blocks of 128 queries.
// ---------------------------------------------------------------------------
__global__ __launch_bounds__(256)
void attention_kernel(const bf16* __restrict__ Q, const bf16* __restrict__ Km,
                      const bf16* __restrict__ V, bf16* __restrict__ Ctx) {
  __shared__ __align__(32) bf16 sP[8][16][32];  // per-wave P transpose buffer

  const int tid  = threadIdx.x;
  const int lane = tid & 31;
  const int w    = tid >> 5;
  const int l16  = lane & 15;
  const int hi   = lane >> 4;
  const int b    = blockIdx.y;
  const size_t boff = (size_t)b * NPIX * CCH;
  Q += boff; Km += boff; V += boff; Ctx += boff;
  const int nq0 = blockIdx.x * 128 + w * 16;

  // Preload this wave's 16 query rows as 8 A-fragments (K = 256).
  v16bf qf[8];
  {
    const bf16* qrow = Q + (size_t)(nq0 + l16) * CCH;
#pragma unroll
    for (int kk = 0; kk < 8; ++kk) {
      const bf16* p = qrow + kk * 32 + hi * 16;
      qf[kk] = load16(p);
    }
  }

  v8f acc[16];
  v8f zero = {};
#pragma unroll
  for (int f = 0; f < 16; ++f) acc[f] = zero;
  float mx[8], li[8];
#pragma unroll
  for (int r = 0; r < 8; ++r) { mx[r] = -1e30f; li[r] = 0.f; }

  for (int m0 = 0; m0 < NPIX; m0 += 32) {
    if (m0 + 32 < NPIX) {  // hint next K chunk toward the caches
      __builtin_prefetch(Km + (size_t)(m0 + 32 + lane) * CCH, 0, 0);
    }
    // Logits: 16(n) x 32(m) tile = two 16x16 f32 fragments, contract K=256.
    v8f s0 = zero, s1 = zero;
    const bf16* kr0 = Km + (size_t)(m0 + l16) * CCH + hi * 16;
    const bf16* kr1 = Km + (size_t)(m0 + 16 + l16) * CCH + hi * 16;
#pragma unroll
    for (int kk = 0; kk < 8; ++kk) {
      s0 = wmma_bf16(qf[kk], load16(kr0 + kk * 32), s0);
      s1 = wmma_bf16(qf[kk], load16(kr1 + kk * 32), s1);
    }

    // Online softmax. Fragment rows live in 16-lane halves -> shfl-xor {8,4,2,1}.
#pragma unroll
    for (int r = 0; r < 8; ++r) {
      float v = fmaxf(s0[r], s1[r]);
#pragma unroll
      for (int off = 8; off >= 1; off >>= 1) v = fmaxf(v, __shfl_xor(v, off, 32));
      float mn    = fmaxf(mx[r], v);
      float alpha = __expf(mx[r] - mn);
      mx[r] = mn;
      li[r] *= alpha;
#pragma unroll
      for (int f = 0; f < 16; ++f) acc[f][r] *= alpha;
      float e0 = __expf(s0[r] - mn);
      float e1 = __expf(s1[r] - mn);
      float sv = e0 + e1;
#pragma unroll
      for (int off = 8; off >= 1; off >>= 1) sv += __shfl_xor(sv, off, 32);
      li[r] += sv;
      // Stash P (bf16) into LDS in [row][col] so we can re-read in A layout.
      sP[w][hi * 8 + r][l16]      = (bf16)e0;
      sP[w][hi * 8 + r][16 + l16] = (bf16)e1;
    }

    // P 16x32 as A-fragment (DS ops are in-order within a wave).
    v16bf pa = load16(&sP[w][l16][hi * 16]);

    // ctx(16 x 256) += P(16x32) * V(32 x 256): 16 column tiles, 1 WMMA each.
#pragma unroll
    for (int kt = 0; kt < 16; ++kt) {
      const bf16* vp = V + (size_t)(kt * 16 + l16) * NPIX + m0 + hi * 16;
      acc[kt] = wmma_bf16(pa, load16(vp), acc[kt]);
    }
  }

  // Normalize and store ctx as bf16 [K, N] (rows of 8 n are contiguous per lane).
  float inv[8];
#pragma unroll
  for (int r = 0; r < 8; ++r) inv[r] = 1.0f / li[r];
#pragma unroll
  for (int kt = 0; kt < 16; ++kt) {
    v8bf ov;
#pragma unroll
    for (int r = 0; r < 8; ++r) ov[r] = (bf16)(acc[kt][r] * inv[r]);
    *(v8bf*)(Ctx + (size_t)(kt * 16 + l16) * NPIX + nq0 + hi * 8) = ov;
  }
}

// ---------------------------------------------------------------------------
// Host launcher
// ---------------------------------------------------------------------------
extern "C" void kernel_launch(void* const* d_in, const int* in_sizes, int n_in,
                              void* d_out, int out_size, void* d_ws, size_t ws_size,
                              hipStream_t stream) {
  const float* x  = (const float*)d_in[0];
  const float* fk = (const float*)d_in[1];
  const float* Ws = (const float*)d_in[2];
  const float* g  = (const float*)d_in[3];
  const float* be = (const float*)d_in[4];
  const float* me = (const float*)d_in[5];
  const float* va = (const float*)d_in[6];
  float* out = (float*)d_out;

  char* ws = (char*)d_ws;
  size_t off = 0;
  auto take = [&](size_t bytes) -> void* {
    void* p = ws + off;
    off += (bytes + 255) & ~(size_t)255;
    return p;
  };
  const size_t WN = 6 * 256 * 256;
  const size_t XN = (size_t)NBATCH * CCH * NPIX;

  bf16*  Wbf   = (bf16*)take(WN * 2);
  float* scale = (float*)take(6 * 256 * 4);
  float* shift = (float*)take(6 * 256 * 4);
  bf16*  xbf   = (bf16*)take(XN * 2);
  bf16*  fbf   = (bf16*)take(XN * 2);
  bf16*  tbf   = (bf16*)take(XN * 2);
  bf16*  qnk   = (bf16*)take(XN * 2);
  bf16*  knk   = (bf16*)take(XN * 2);
  bf16*  vkn   = (bf16*)take(XN * 2);
  bf16*  ctx   = (bf16*)take(XN * 2);
  (void)ws_size; (void)in_sizes; (void)n_in; (void)out_size;

  prep_kernel<<<dim3((unsigned)((WN + 255) / 256)), 256, 0, stream>>>(
      Ws, g, be, me, va, Wbf, scale, shift);
  cvt_kernel<<<dim3((unsigned)(XN / 256)), 256, 0, stream>>>(x, xbf, (int)XN);
  cvt_kernel<<<dim3((unsigned)(XN / 256)), 256, 0, stream>>>(fk, fbf, (int)XN);

  dim3 cg(NPIX / 64, CCH / 32, NBATCH);
  // psi: two convbn+relu -> q [N,K]
  convbn_kernel<0><<<cg, 256, 0, stream>>>(xbf, Wbf + 0 * 65536, scale + 0,    shift + 0,    tbf, nullptr);
  convbn_kernel<1><<<cg, 256, 0, stream>>>(tbf, Wbf + 1 * 65536, scale + 256,  shift + 256,  qnk, nullptr);
  // phi: two convbn+relu -> k [N,K] (transposed at the store)
  convbn_kernel<0><<<cg, 256, 0, stream>>>(fbf, Wbf + 2 * 65536, scale + 512,  shift + 512,  tbf, nullptr);
  convbn_kernel<1><<<cg, 256, 0, stream>>>(tbf, Wbf + 3 * 65536, scale + 768,  shift + 768,  knk, nullptr);
  // f_down -> v [K,N]
  convbn_kernel<0><<<cg, 256, 0, stream>>>(fbf, Wbf + 4 * 65536, scale + 1024, shift + 1024, vkn, nullptr);

  attention_kernel<<<dim3(NPIX / 128, NBATCH), 256, 0, stream>>>(qnk, knk, vkn, ctx);

  // f_up -> f32 NCHW output
  convbn_kernel<2><<<cg, 256, 0, stream>>>(ctx, Wbf + 5 * 65536, scale + 1280, shift + 1280, nullptr, out);
}
